// UniversalKnowledgeIntegrator_50568944943393
// MI455X (gfx1250) — matrix operand order repaired and checked by reading, not compiled, not run on hardware
//
#include <hip/hip_runtime.h>
#include <hip/hip_bf16.h>

// ---------------------------------------------------------------------------
// UniversalKnowledgeIntegrator for MI455X (gfx1250, wave32, WMMA, async-LDS)
//
//   k_mean_convert : xm = mean(x, s), xb = bf16(x)
//   k_wconv        : w1aT = bf16(i_w1[:1024].T), w2T = bf16(i_w2.T)
//   k_r1           : h = relu(xm @ r_w1 + r_b1)
//   k_logits       : logits = h @ r_w2 + r_b2      (memory-bound, 16 rows)
//   k_softmax      : weights = softmax(logits)     (in place)
//   k_wk           : wk = weights @ emb            (split-K + f32 atomics)
//   k_wkc          : wkc = wk @ i_w1[1024:] + i_b1 (per-batch bias, tiny)
//   k_fused_mlp    : out = tanh(gelu(xb @ w1aT^T + wkc) @ w2T^T + i_b2)
//        - two back-to-back bf16 WMMA GEMMs per 128-row block, each stage
//          fully specialized at compile time (template<int STAGE>)
//        - intermediate G lives in LDS (128 x 1032 bf16 = 258KB)
//        - weight tiles staged once per WG via global_load_async_to_lds_b128
//          (double-buffered, ASYNCcnt/s_wait_asynccnt), shared by all 8 waves
//        - total LDS: 294KB of the 320KB CDNA5 WGP budget
// ---------------------------------------------------------------------------

typedef __attribute__((ext_vector_type(16))) __bf16 v16bf;
typedef __attribute__((ext_vector_type(8)))  float  v8f;

#define VOCAB 100000
#define DMODEL 1024
#define NB 16
#define NS 2048
#define NROWS (NB * NS)      // 32768
#define GSTRIDE 1032         // bf16 elems per G LDS row (516 dwords -> +4 banks/row)
#define BSTRIDE 72           // bf16 elems per B LDS row (36 dwords -> conflict-free b128)
#define BCHUNK_ELEMS (128 * BSTRIDE)        // one 128N x 64K staged chunk (padded)
#define GS_ELEMS (128 * GSTRIDE)

__device__ __forceinline__ unsigned short f2bf(float f) {
    unsigned u = __float_as_uint(f);
    unsigned r = u + 0x7FFFu + ((u >> 16) & 1u);   // round-to-nearest-even
    return (unsigned short)(r >> 16);
}

union FragBF {
    uint4 u[2];
    v16bf v;
};

// CDNA5 async copy: global -> LDS, tracked by ASYNCcnt
__device__ __forceinline__ void async_b128(unsigned lds_byte_addr, const void* gptr) {
    asm volatile("global_load_async_to_lds_b128 %0, %1, off"
                 :: "v"(lds_byte_addr), "v"(gptr)
                 : "memory");
}
__device__ __forceinline__ void wait_async_le4() {
    asm volatile("s_wait_asynccnt 4" ::: "memory");
}
__device__ __forceinline__ void wait_async_le0() {
    asm volatile("s_wait_asynccnt 0" ::: "memory");
}

// ---------------------------------------------------------------- mean + bf16
__global__ void k_mean_convert(const float* __restrict__ x,
                               float* __restrict__ xm,
                               unsigned short* __restrict__ xb) {
    int idx = blockIdx.x * blockDim.x + threadIdx.x;   // 0..16383
    int b = idx >> 10, d = idx & 1023;
    const float* px = x + ((size_t)b * NS) * DMODEL + d;
    float sum = 0.f;
    for (int s = 0; s < NS; ++s) {
        float v = px[(size_t)s * DMODEL];
        sum += v;
        xb[((size_t)b * NS + s) * DMODEL + d] = f2bf(v);
    }
    xm[idx] = sum * (1.0f / NS);
}

// ----------------------------------------------------- weight convert+transpose
__global__ void k_wconv(const float* __restrict__ i_w1,
                        const float* __restrict__ i_w2,
                        unsigned short* __restrict__ w1aT,
                        unsigned short* __restrict__ w2T) {
    int idx = blockIdx.x * blockDim.x + threadIdx.x;   // 0..2M-1
    int sel = idx >> 20;
    int e = idx & ((1 << 20) - 1);
    int k = e >> 10, n = e & 1023;
    if (sel == 0) w1aT[(size_t)n * DMODEL + k] = f2bf(i_w1[(size_t)k * DMODEL + n]);
    else          w2T [(size_t)n * DMODEL + k] = f2bf(i_w2[(size_t)k * DMODEL + n]);
}

// ---------------------------------------------------------------- router fc1
__global__ void k_r1(const float* __restrict__ xm,
                     const float* __restrict__ r_w1,
                     const float* __restrict__ r_b1,
                     float* __restrict__ h) {
    int idx = blockIdx.x * blockDim.x + threadIdx.x;   // 0..8191
    int b = idx >> 9, j = idx & 511;
    float s = r_b1[j];
    for (int k = 0; k < DMODEL; ++k)
        s += xm[b * DMODEL + k] * r_w1[(size_t)k * 512 + j];
    h[idx] = fmaxf(s, 0.f);
}

// ---------------------------------------------------------------- router fc2
__global__ void k_logits(const float* __restrict__ h,
                         const float* __restrict__ r_w2,
                         const float* __restrict__ r_b2,
                         float* __restrict__ logits) {
    __shared__ float hs[NB * 512];
    for (int i = threadIdx.x; i < NB * 512; i += blockDim.x) hs[i] = h[i];
    __syncthreads();
    int n = blockIdx.x * blockDim.x + threadIdx.x;
    if (n >= VOCAB) return;
    float acc[NB];
    float bias = r_b2[n];
#pragma unroll
    for (int b = 0; b < NB; ++b) acc[b] = bias;
    for (int k = 0; k < 512; ++k) {
        float w = r_w2[(size_t)k * VOCAB + n];
#pragma unroll
        for (int b = 0; b < NB; ++b) acc[b] += hs[b * 512 + k] * w;
    }
#pragma unroll
    for (int b = 0; b < NB; ++b) logits[(size_t)b * VOCAB + n] = acc[b];
}

// ---------------------------------------------------------------- softmax rows
__global__ void k_softmax(float* __restrict__ logits) {
    int b = blockIdx.x, tid = threadIdx.x;
    float* row = logits + (size_t)b * VOCAB;
    __shared__ float red[256];
    float m = -3.4e38f;
    for (int i = tid; i < VOCAB; i += 256) m = fmaxf(m, row[i]);
    red[tid] = m; __syncthreads();
    for (int o = 128; o > 0; o >>= 1) {
        if (tid < o) red[tid] = fmaxf(red[tid], red[tid + o]);
        __syncthreads();
    }
    m = red[0]; __syncthreads();
    float s = 0.f;
    for (int i = tid; i < VOCAB; i += 256) s += expf(row[i] - m);
    red[tid] = s; __syncthreads();
    for (int o = 128; o > 0; o >>= 1) {
        if (tid < o) red[tid] += red[tid + o];
        __syncthreads();
    }
    float inv = 1.0f / red[0];
    for (int i = tid; i < VOCAB; i += 256) row[i] = expf(row[i] - m) * inv;
}

// ------------------------------------------------------- wk = weights @ emb
#define VCHUNK 1024
__global__ void k_wk(const float* __restrict__ weights,
                     const float* __restrict__ emb,
                     float* __restrict__ wk) {
    int d = blockIdx.x * blockDim.x + threadIdx.x;     // 0..1023
    int v0 = blockIdx.y * VCHUNK;
    int v1 = min(VOCAB, v0 + VCHUNK);
    float acc[NB];
#pragma unroll
    for (int b = 0; b < NB; ++b) acc[b] = 0.f;
    for (int v = v0; v < v1; ++v) {
        float e = emb[(size_t)v * DMODEL + d];
#pragma unroll
        for (int b = 0; b < NB; ++b) acc[b] += weights[(size_t)b * VOCAB + v] * e;
    }
#pragma unroll
    for (int b = 0; b < NB; ++b) atomicAdd(&wk[b * DMODEL + d], acc[b]);
}

// -------------------------------------- wkc = wk @ i_w1[1024:] + i_b1 (bias)
__global__ void k_wkc(const float* __restrict__ wk,
                      const float* __restrict__ i_w1,
                      const float* __restrict__ i_b1,
                      float* __restrict__ wkc) {
    int idx = blockIdx.x * blockDim.x + threadIdx.x;   // 0..16383
    int b = idx >> 10, n = idx & 1023;
    float s = i_b1[n];
    for (int k = 0; k < DMODEL; ++k)
        s += wk[b * DMODEL + k] * i_w1[(size_t)(DMODEL + k) * DMODEL + n];
    wkc[idx] = s;
}

// ---------------------------------------------------------------------------
// One GEMM stage, fully specialized at compile time.
//   STAGE==0: A from global (xb rows), epilogue = gelu -> Gs (LDS, bf16)
//   STAGE==1: A from Gs (LDS),         epilogue = tanh -> out (global, f32)
// B (wT) is async-staged: per pass p, K consumed in 16 chunks of 64; each
// 128N x 64K bf16 chunk copied to LDS once, consumed by all 8 waves.
// ---------------------------------------------------------------------------
template <int STAGE>
__device__ __forceinline__ void gemm_stage(
    unsigned short* Gs, unsigned short* Bs, unsigned bsAddr0,
    const unsigned short* __restrict__ aRow,   // STAGE==0 A source (this lane's row)
    const unsigned short* __restrict__ wT,     // B matrix, [N][K] bf16
    const float* __restrict__ biasv,           // stage bias vector base (len 1024)
    float* __restrict__ out, int blockRow0,
    int tid, int lrow0, int half, int mr) {
    for (int p = 0; p < 8; ++p) {
        const int n0 = p * 128;
        v8f acc[8];
#pragma unroll
        for (int t = 0; t < 8; ++t) {
            float bias = biasv[n0 + t * 16 + mr];
#pragma unroll
            for (int i = 0; i < 8; ++i) acc[t][i] = bias;
        }
        for (int c = 0; c < 16; ++c) {
            __syncthreads();   // readers of chunk c-1 (in buffer (c+1)&1) done
#pragma unroll
            for (int pre = 0; pre < 2; ++pre) {     // c==0 primes buffer 0 too
                int cc = c + ((c == 0) ? pre : 1);
                if ((c == 0 || pre == 1) && cc < 16) {
                    unsigned base = bsAddr0 + (unsigned)(cc & 1) * (BCHUNK_ELEMS * 2);
#pragma unroll
                    for (int r = 0; r < 4; ++r) {
                        int li = tid + 256 * r;          // 0..1023
                        int n  = li >> 3;                // 0..127
                        int j  = li & 7;                 // 16B segment in 64K row
                        const unsigned short* g =
                            wT + (size_t)(n0 + n) * DMODEL + cc * 64 + j * 8;
                        async_b128(base + (unsigned)(n * BSTRIDE + j * 8) * 2, g);
                    }
                }
            }
            if (c < 15) wait_async_le4(); else wait_async_le0();
            __syncthreads();   // chunk c resident in LDS for all waves
            const unsigned short* bBuf = Bs + (c & 1) * BCHUNK_ELEMS;
#pragma unroll
            for (int kh = 0; kh < 2; ++kh) {
                const int kb   = c * 64 + kh * 32;   // global K
                const int kloc = kh * 32;            // K within staged chunk
                FragBF a;
                if (STAGE == 0) {
                    a.u[0] = *(const uint4*)(aRow + kb + half * 8);
                    a.u[1] = *(const uint4*)(aRow + kb + 16 + half * 8);
                    __builtin_prefetch(aRow + kb + 64, 0, 3);
                } else {
                    const unsigned short* gRow = &Gs[(lrow0 + mr) * GSTRIDE + kb];
                    a.u[0] = *(const uint4*)(gRow + half * 8);
                    a.u[1] = *(const uint4*)(gRow + 16 + half * 8);
                }
#pragma unroll
                for (int t = 0; t < 8; ++t) {
                    const unsigned short* bp =
                        bBuf + (t * 16 + mr) * BSTRIDE + kloc + half * 8;
                    FragBF bm;
                    bm.u[0] = *(const uint4*)(bp);
                    bm.u[1] = *(const uint4*)(bp + 16);
                    acc[t] = __builtin_amdgcn_wmma_f32_16x16x32_bf16(
                        false, a.v, false, bm.v, (short)0, acc[t], false, false);
                }
            }
        }
        // epilogue
#pragma unroll
        for (int t = 0; t < 8; ++t) {
#pragma unroll
            for (int i = 0; i < 8; ++i) {
                int M = lrow0 + i + half * 8;
                int N = n0 + t * 16 + mr;
                if (STAGE == 0) {
                    float g = acc[t][i];
                    g = 0.5f * g * (1.0f + erff(g * 0.70710678118654752f));
                    Gs[M * GSTRIDE + N] = f2bf(g);
                } else {
                    out[(size_t)(blockRow0 + M) * DMODEL + N] = tanhf(acc[t][i]);
                }
            }
        }
    }
}

__global__ __launch_bounds__(256, 1) void k_fused_mlp(
    const unsigned short* __restrict__ xb,
    const unsigned short* __restrict__ w1aT,
    const unsigned short* __restrict__ w2T,
    const float* __restrict__ wkc,
    const float* __restrict__ i_b2,
    float* __restrict__ out) {
    extern __shared__ unsigned short Gs[];             // [GS_ELEMS] G + [2*BCHUNK] B
    unsigned short* Bs = Gs + GS_ELEMS;
    const unsigned bsAddr0 = (unsigned)(unsigned long long)(&Bs[0]);  // LDS byte addr

    const int tid   = threadIdx.x;
    const int lane  = tid & 31;
    const int half  = lane >> 4;
    const int mr    = lane & 15;
    const int lrow0 = (tid >> 5) * 16;
    const int blockRow0 = blockIdx.x * 128;
    const int batch = blockRow0 >> 11;

    const unsigned short* aRow = xb + (size_t)(blockRow0 + lrow0 + mr) * DMODEL;

    gemm_stage<0>(Gs, Bs, bsAddr0, aRow, w1aT, wkc + batch * DMODEL,
                  out, blockRow0, tid, lrow0, half, mr);
    __syncthreads();   // G fully written before stage 2 consumes it
    gemm_stage<1>(Gs, Bs, bsAddr0, aRow, w2T, i_b2,
                  out, blockRow0, tid, lrow0, half, mr);
}

// ---------------------------------------------------------------------------
extern "C" void kernel_launch(void* const* d_in, const int* in_sizes, int n_in,
                              void* d_out, int out_size, void* d_ws, size_t ws_size,
                              hipStream_t stream) {
    const float* x    = (const float*)d_in[0];
    const float* emb  = (const float*)d_in[1];
    const float* r_w1 = (const float*)d_in[2];
    const float* r_b1 = (const float*)d_in[3];
    const float* r_w2 = (const float*)d_in[4];
    const float* r_b2 = (const float*)d_in[5];
    const float* i_w1 = (const float*)d_in[6];
    const float* i_b1 = (const float*)d_in[7];
    const float* i_w2 = (const float*)d_in[8];
    const float* i_b2 = (const float*)d_in[9];
    float* out = (float*)d_out;

    char* ws = (char*)d_ws;
    size_t off = 0;
    auto alloc = [&](size_t bytes) {
        char* p = ws + off;
        off = (off + bytes + 255) & ~(size_t)255;
        return p;
    };
    unsigned short* xb   = (unsigned short*)alloc((size_t)NROWS * DMODEL * 2);
    float* xm            = (float*)alloc((size_t)NB * DMODEL * 4);
    float* h             = (float*)alloc((size_t)NB * 512 * 4);
    float* logits        = (float*)alloc((size_t)NB * VOCAB * 4);
    float* wk            = (float*)alloc((size_t)NB * DMODEL * 4);
    float* wkc           = (float*)alloc((size_t)NB * DMODEL * 4);
    unsigned short* w1aT = (unsigned short*)alloc((size_t)DMODEL * DMODEL * 2);
    unsigned short* w2T  = (unsigned short*)alloc((size_t)DMODEL * DMODEL * 2);

    hipMemsetAsync(wk, 0, (size_t)NB * DMODEL * 4, stream);

    k_mean_convert<<<(NB * DMODEL) / 256, 256, 0, stream>>>(x, xm, xb);
    k_wconv<<<(2 * DMODEL * DMODEL) / 256, 256, 0, stream>>>(i_w1, i_w2, w1aT, w2T);
    k_r1<<<(NB * 512) / 256, 256, 0, stream>>>(xm, r_w1, r_b1, h);
    k_logits<<<(VOCAB + 255) / 256, 256, 0, stream>>>(h, r_w2, r_b2, logits);
    k_softmax<<<NB, 256, 0, stream>>>(logits);
    k_wk<<<dim3(DMODEL / 256, (VOCAB + VCHUNK - 1) / VCHUNK), 256, 0, stream>>>(logits, emb, wk);
    k_wkc<<<(NB * DMODEL) / 256, 256, 0, stream>>>(wk, i_w1, i_b1, wkc);

    const int lds_bytes = (GS_ELEMS + 2 * BCHUNK_ELEMS) * 2;   // 301056 B <= 320KB WGP
    hipFuncSetAttribute((const void*)k_fused_mlp,
                        hipFuncAttributeMaxDynamicSharedMemorySize, lds_bytes);
    k_fused_mlp<<<NROWS / 128, 256, lds_bytes, stream>>>(xb, w1aT, w2T, wkc, i_b2, out);
}
